// MultiHeadAttention_59622736003936
// MI455X (gfx1250) — matrix operand order repaired and checked by reading, not compiled, not run on hardware
//
#include <hip/hip_runtime.h>
#include <math.h>

// Problem constants (from reference)
#define BB 4
#define TT 2048
#define DD 1024
#define HH 16
#define DH 64
#define BT (BB * TT)

typedef __attribute__((ext_vector_type(16))) __bf16 v16bf;
typedef __attribute__((ext_vector_type(8)))  __bf16 v8bf;
typedef __attribute__((ext_vector_type(8)))  float  v8f;

union Frag16 { v16bf v; v8bf h[2]; };

// A-matrix 16x32 bf16 fragment: lane l holds row m = l&15,
// K = 16*(l>>4) + 0..15 contiguous -> two b128 loads.
static __device__ __forceinline__ v16bf load_a_frag(const __bf16* base, int row0,
                                                    int k0, int ld, int lane) {
  const __bf16* p = base + (long)(row0 + (lane & 15)) * ld + k0 + ((lane >> 4) << 4);
  Frag16 f;
  f.h[0] = *(const v8bf*)(p);
  f.h[1] = *(const v8bf*)(p + 8);
  return f.v;
}

// B-matrix 32x16 bf16 fragment: lane l holds column n = l&15,
// K = 16*(l>>4) + 0..15 contiguous -> two b128 loads. Source must be stored
// column-major of B, i.e. (N x K) row-major ("Bt").
static __device__ __forceinline__ v16bf load_b_frag(const __bf16* base, int col0,
                                                    int k0, int ld, int lane) {
  const __bf16* p = base + (long)(col0 + (lane & 15)) * ld + k0 + ((lane >> 4) << 4);
  Frag16 f;
  f.h[0] = *(const v8bf*)(p);
  f.h[1] = *(const v8bf*)(p + 8);
  return f.v;
}

static __device__ __forceinline__ v8f wmma_bf16(v16bf a, v16bf b, v8f c) {
  // D = A*B + C ; v_wmma_f32_16x16x32_bf16
  return __builtin_amdgcn_wmma_f32_16x16x32_bf16(false, a, false, b, (short)0, c,
                                                 false, false);
}

// ---------------------------------------------------------------------------
// Kernel 0: convert x to bf16; transpose+convert the four weights to (N x K).
// ---------------------------------------------------------------------------
__global__ void prep_kernel(const float* __restrict__ x,
                            const float* __restrict__ wq, const float* __restrict__ wk,
                            const float* __restrict__ wv, const float* __restrict__ wo,
                            __bf16* __restrict__ xb,
                            __bf16* __restrict__ wqt, __bf16* __restrict__ wkt,
                            __bf16* __restrict__ wvt, __bf16* __restrict__ wot) {
  long stride = (long)gridDim.x * blockDim.x;
  long tid = (long)blockIdx.x * blockDim.x + threadIdx.x;
  for (long i = tid; i < (long)BT * DD; i += stride) xb[i] = (__bf16)x[i];
  for (long i = tid; i < (long)DD * DD; i += stride) {
    long n = i / DD, kk = i - n * DD;
    long src = kk * DD + n;            // wt[n][k] = w[k][n]
    wqt[i] = (__bf16)wq[src];
    wkt[i] = (__bf16)wk[src];
    wvt[i] = (__bf16)wv[src];
    wot[i] = (__bf16)wo[src];
  }
}

// ---------------------------------------------------------------------------
// Kernel 1: C(M x N) = A(M x K) * B(K x N), bf16 in / bf16 out.
// Each wave computes a 32x64 tile: 2 A-fragments x 4 B-fragments = 8 WMMAs
// per 32-wide k-step (12 b128 loads -> ~40% less load traffic per WMMA than
// a 16x64 tile).
// vmode 0: row-major (BT x DD) store; vmode 1: store V transposed (B,H,Dh,T).
// ---------------------------------------------------------------------------
__global__ void __launch_bounds__(128) gemm_bf16_kernel(const __bf16* __restrict__ A,
                                                        const __bf16* __restrict__ Bt,
                                                        __bf16* __restrict__ out,
                                                        int vmode) {
  int lane = threadIdx.x & 31;
  int wave = threadIdx.x >> 5;
  int m0 = blockIdx.y * 32;
  int n0 = (blockIdx.x * 4 + wave) * 64;

  v8f acc[2][4];
#pragma unroll
  for (int i = 0; i < 2; ++i)
#pragma unroll
    for (int j = 0; j < 4; ++j)
#pragma unroll
      for (int r = 0; r < 8; ++r) acc[i][j][r] = 0.0f;

  for (int k = 0; k < DD; k += 32) {
    v16bf a0 = load_a_frag(A, m0, k, DD, lane);
    v16bf a1 = load_a_frag(A, m0 + 16, k, DD, lane);
#pragma unroll
    for (int j = 0; j < 4; ++j) {
      v16bf b = load_b_frag(Bt, n0 + 16 * j, k, DD, lane);
      acc[0][j] = wmma_bf16(a0, b, acc[0][j]);
      acc[1][j] = wmma_bf16(a1, b, acc[1][j]);
    }
  }

  int ncol = lane & 15;
  int rb = (lane >> 4) << 3;            // C layout: M = rb + r, N = ncol
  if (vmode == 0) {
#pragma unroll
    for (int i = 0; i < 2; ++i)
#pragma unroll
      for (int j = 0; j < 4; ++j)
#pragma unroll
        for (int r = 0; r < 8; ++r) {
          int m = m0 + 16 * i + rb + r, n = n0 + 16 * j + ncol;
          out[(long)m * DD + n] = (__bf16)acc[i][j][r];
        }
  } else {
#pragma unroll
    for (int i = 0; i < 2; ++i)
#pragma unroll
      for (int j = 0; j < 4; ++j)
#pragma unroll
        for (int r = 0; r < 8; ++r) {
          int m = m0 + 16 * i + rb + r, n = n0 + 16 * j + ncol;
          int b = m >> 11, t = m & (TT - 1);
          int h = n >> 6, dh = n & (DH - 1);
          out[(((long)(b * HH + h)) * DH + dh) * TT + t] = (__bf16)acc[i][j][r];
        }
  }
}

// ---------------------------------------------------------------------------
// Kernel 2: causal flash attention. One wave handles 16 queries of one (b,h).
// Computes S^T = K * Q^T (so each lane owns one query column), online softmax
// with one shfl_xor(16) per statistic, ctx^T += V^T * P via WMMA.
// ---------------------------------------------------------------------------
__global__ void __launch_bounds__(128) attn_kernel(const __bf16* __restrict__ q,
                                                   const __bf16* __restrict__ kmat,
                                                   const __bf16* __restrict__ vt,
                                                   __bf16* __restrict__ ctxb) {
  int lane = threadIdx.x & 31;
  int wave = threadIdx.x >> 5;
  int t0 = (blockIdx.x * 4 + wave) * 16;     // query tile start
  int bh = blockIdx.y;
  int b = bh >> 4, h = bh & (HH - 1);

  const __bf16* qbase = q + (long)b * TT * DD + h * DH;    // (T x Dh), ld = DD
  const __bf16* kbase = kmat + (long)b * TT * DD + h * DH; // (T x Dh), ld = DD
  const __bf16* vbase = vt + (long)bh * DH * TT;           // (Dh x T), ld = TT

  // Q^T as B-fragments (column n = query, k = dh), for both dh halves.
  v16bf qf0 = load_b_frag(qbase, t0, 0, DD, lane);
  v16bf qf1 = load_b_frag(qbase, t0, 32, DD, lane);

  v8f c0, c1, c2, c3;                        // ctx^T: dh tiles 0..3, N = query
#pragma unroll
  for (int r = 0; r < 8; ++r) { c0[r] = 0.f; c1[r] = 0.f; c2[r] = 0.f; c3[r] = 0.f; }

  float mrun = -INFINITY, lrun = 0.0f;
  int qidx = t0 + (lane & 15);
  int rb = (lane >> 4) << 3;
  bool hihalf = (lane >= 16);

  int kend = t0 + 16;                        // exclusive key bound (causal)
  for (int kc = 0; kc < kend; kc += 32) {
    // S^T fragments for the two 16-key subtiles (rows M = key index).
    v8f s0, s1;
#pragma unroll
    for (int r = 0; r < 8; ++r) { s0[r] = 0.f; s1[r] = 0.f; }
    s0 = wmma_bf16(load_a_frag(kbase, kc, 0, DD, lane), qf0, s0);
    s0 = wmma_bf16(load_a_frag(kbase, kc, 32, DD, lane), qf1, s0);
    s1 = wmma_bf16(load_a_frag(kbase, kc + 16, 0, DD, lane), qf0, s1);
    s1 = wmma_bf16(load_a_frag(kbase, kc + 16, 32, DD, lane), qf1, s1);

    // Scale by 1/sqrt(64) and apply causal mask analytically.
    float sv0[8], sv1[8];
    float cmax = -INFINITY;
#pragma unroll
    for (int r = 0; r < 8; ++r) {
      int key0 = kc + rb + r;
      int key1 = kc + 16 + rb + r;
      sv0[r] = (key0 <= qidx) ? s0[r] * 0.125f : -INFINITY;
      sv1[r] = (key1 <= qidx) ? s1[r] * 0.125f : -INFINITY;
      cmax = fmaxf(cmax, fmaxf(sv0[r], sv1[r]));
    }
    cmax = fmaxf(cmax, __shfl_xor(cmax, 16, 32));   // combine lane halves
    float mnew = fmaxf(mrun, cmax);
    float alpha = __expf(mrun - mnew);

    float p0[8], p1[8], psum = 0.0f;
#pragma unroll
    for (int r = 0; r < 8; ++r) {
      p0[r] = __expf(sv0[r] - mnew);
      p1[r] = __expf(sv1[r] - mnew);
      psum += p0[r] + p1[r];
    }
    psum += __shfl_xor(psum, 16, 32);
    lrun = lrun * alpha + psum;
    mrun = mnew;

    // Repack P (C layout, split 8+8 across lane halves) into a 32x16
    // B-fragment (low lanes: K 0..15, high lanes: K 16..31).
    Frag16 pf;
#pragma unroll
    for (int r = 0; r < 8; ++r) {
      float o0 = __shfl_xor(p0[r], 16, 32);
      float o1 = __shfl_xor(p1[r], 16, 32);
      pf.v[r]     = (__bf16)(hihalf ? o1 : p0[r]);   // element r
      pf.v[8 + r] = (__bf16)(hihalf ? p1[r] : o0);   // element 8+r
    }

#pragma unroll
    for (int r = 0; r < 8; ++r) {
      c0[r] *= alpha; c1[r] *= alpha; c2[r] *= alpha; c3[r] *= alpha;
    }
    // ctx^T += V^T(16dh x 32keys) * P(32keys x 16q)
    c0 = wmma_bf16(load_a_frag(vbase, 0,  kc, TT, lane), pf.v, c0);
    c1 = wmma_bf16(load_a_frag(vbase, 16, kc, TT, lane), pf.v, c1);
    c2 = wmma_bf16(load_a_frag(vbase, 32, kc, TT, lane), pf.v, c2);
    c3 = wmma_bf16(load_a_frag(vbase, 48, kc, TT, lane), pf.v, c3);
  }

  float inv = 1.0f / lrun;
  __bf16* cb = ctxb + (long)b * TT * DD + h * DH;  // ctx row-major (T x D)
  long rowoff = (long)qidx * DD;
#pragma unroll
  for (int r = 0; r < 8; ++r) {
    cb[rowoff + (0  + rb + r)] = (__bf16)(c0[r] * inv);
    cb[rowoff + (16 + rb + r)] = (__bf16)(c1[r] * inv);
    cb[rowoff + (32 + rb + r)] = (__bf16)(c2[r] * inv);
    cb[rowoff + (48 + rb + r)] = (__bf16)(c3[r] * inv);
  }
}

// ---------------------------------------------------------------------------
// Kernel 3: out = ctx @ wo + bo, f32 store to d_out.  32x64 tile per wave.
// ---------------------------------------------------------------------------
__global__ void __launch_bounds__(128) proj_kernel(const __bf16* __restrict__ A,
                                                   const __bf16* __restrict__ Bt,
                                                   const float* __restrict__ bias,
                                                   float* __restrict__ out) {
  int lane = threadIdx.x & 31;
  int wave = threadIdx.x >> 5;
  int m0 = blockIdx.y * 32;
  int n0 = (blockIdx.x * 4 + wave) * 64;

  v8f acc[2][4];
#pragma unroll
  for (int i = 0; i < 2; ++i)
#pragma unroll
    for (int j = 0; j < 4; ++j)
#pragma unroll
      for (int r = 0; r < 8; ++r) acc[i][j][r] = 0.0f;

  for (int k = 0; k < DD; k += 32) {
    v16bf a0 = load_a_frag(A, m0, k, DD, lane);
    v16bf a1 = load_a_frag(A, m0 + 16, k, DD, lane);
#pragma unroll
    for (int j = 0; j < 4; ++j) {
      v16bf b = load_b_frag(Bt, n0 + 16 * j, k, DD, lane);
      acc[0][j] = wmma_bf16(a0, b, acc[0][j]);
      acc[1][j] = wmma_bf16(a1, b, acc[1][j]);
    }
  }

  int ncol = lane & 15;
  int rb = (lane >> 4) << 3;
#pragma unroll
  for (int i = 0; i < 2; ++i)
#pragma unroll
    for (int j = 0; j < 4; ++j)
#pragma unroll
      for (int r = 0; r < 8; ++r) {
        int m = m0 + 16 * i + rb + r, n = n0 + 16 * j + ncol;
        out[(long)m * DD + n] = acc[i][j][r] + bias[n];
      }
}

// ---------------------------------------------------------------------------
extern "C" void kernel_launch(void* const* d_in, const int* in_sizes, int n_in,
                              void* d_out, int out_size, void* d_ws, size_t ws_size,
                              hipStream_t stream) {
  (void)in_sizes; (void)n_in; (void)out_size; (void)ws_size;
  const float* x  = (const float*)d_in[0];
  const float* wq = (const float*)d_in[1];
  const float* wk = (const float*)d_in[2];
  const float* wv = (const float*)d_in[3];
  const float* wo = (const float*)d_in[4];
  const float* bo = (const float*)d_in[5];
  float* out = (float*)d_out;

  char* ws = (char*)d_ws;
  size_t off = 0;
  auto take = [&](size_t bytes) { char* p = ws + off; off += bytes; return p; };
  __bf16* xb  = (__bf16*)take((size_t)BT * DD * 2);   // 16 MB
  __bf16* wqt = (__bf16*)take((size_t)DD * DD * 2);   // 2 MB
  __bf16* wkt = (__bf16*)take((size_t)DD * DD * 2);
  __bf16* wvt = (__bf16*)take((size_t)DD * DD * 2);
  __bf16* wot = (__bf16*)take((size_t)DD * DD * 2);
  __bf16* qb  = (__bf16*)take((size_t)BT * DD * 2);   // 16 MB
  __bf16* kb  = (__bf16*)take((size_t)BT * DD * 2);   // 16 MB
  __bf16* vtb = (__bf16*)take((size_t)BT * DD * 2);   // 16 MB (B,H,Dh,T)
  __bf16* cxb = (__bf16*)take((size_t)BT * DD * 2);   // 16 MB

  prep_kernel<<<2048, 256, 0, stream>>>(x, wq, wk, wv, wo, xb, wqt, wkt, wvt, wot);

  dim3 gg(DD / 256, BT / 32);   // (4, 256), 4 waves/block, 32x64 tile/wave
  gemm_bf16_kernel<<<gg, 128, 0, stream>>>(xb, wqt, qb, 0);
  gemm_bf16_kernel<<<gg, 128, 0, stream>>>(xb, wkt, kb, 0);
  gemm_bf16_kernel<<<gg, 128, 0, stream>>>(xb, wvt, vtb, 1);

  attn_kernel<<<dim3(TT / 64, BB * HH), 128, 0, stream>>>(qb, kb, vtb, cxb);

  proj_kernel<<<gg, 128, 0, stream>>>(cxb, wot, bo, out);
}
